// RegionProposalNetwork_52810917872159
// MI455X (gfx1250) — matrix operand order repaired and checked by reading, not compiled, not run on hardware
//
#include <hip/hip_runtime.h>

// ---------- types ----------
typedef __attribute__((ext_vector_type(16))) __bf16 v16bf;
typedef __attribute__((ext_vector_type(8)))  float  v8f;

union Frag { v16bf v; uint4 q[2]; };

__device__ __forceinline__ unsigned short f2bf(float f) {
  union { float f; unsigned u; } v; v.f = f;
  unsigned r = v.u + 0x7FFFu + ((v.u >> 16) & 1u);   // round-to-nearest-even
  return (unsigned short)(r >> 16);
}

__device__ __forceinline__ v8f wmma_bf16(const Frag& a, const Frag& b, v8f c) {
  return __builtin_amdgcn_wmma_f32_16x16x32_bf16(false, a.v, false, b.v,
                                                 (short)0, c, false, false);
}

// ---------- async global->LDS (gfx1250 ASYNCcnt path, inline asm) ----------
// global_load_async_to_lds_b128: VDST = LDS byte-offset VGPR, VADDR = 64-bit addr pair.
// Generic __shared__ pointers are aperture|offset; low 32 bits are the LDS offset
// so a simple truncation gives the VDST value.
__device__ __forceinline__ void async_copy16(const void* g, void* l) {
  unsigned loff = (unsigned)(unsigned long long)l;
  unsigned long long ga = (unsigned long long)g;
  asm volatile("global_load_async_to_lds_b128 %0, %1, off"
               :: "v"(loff), "v"(ga) : "memory");
}

#if defined(__has_builtin)
#if __has_builtin(__builtin_amdgcn_s_wait_asynccnt)
#define ASYNCWAIT_BUILTIN 1
#endif
#endif

template <int N>
__device__ __forceinline__ void wait_async() {
#ifdef ASYNCWAIT_BUILTIN
  __builtin_amdgcn_s_wait_asynccnt(N);
#else
  asm volatile("s_wait_asynccnt %0" :: "i"(N) : "memory");
#endif
}

// ---------- constants ----------
#define HW    10000     // 100x100 pixels
#define CIN   512
#define K3    4608      // 9 * 512
#define PADW  102       // zero-padded spatial width
#define NANC  9
#define NPROP 90000     // HW * NANC
#define PRE_K 2000
#define POST_K 1000
#define NMSW  63        // ceil(2000/32)
#define KSTEPS 144      // K3 / 32

// ================= pack kernels =================
__global__ void zero_pad_k(unsigned* __restrict__ featP32) {
  int gid = blockIdx.x * 256 + threadIdx.x;
  if (gid < (PADW * PADW * CIN) / 2) featP32[gid] = 0u;
}

__global__ void pack_feat_k(const float* __restrict__ feat, unsigned short* __restrict__ featP) {
  int gid = blockIdx.x * 256 + threadIdx.x;
  if (gid >= HW * CIN) return;
  int p = gid / CIN, c = gid % CIN;
  int y = p / 100, x = p % 100;
  featP[((y + 1) * PADW + (x + 1)) * CIN + c] = f2bf(feat[c * HW + p]);
}

__global__ void pack_wconv_k(const float* __restrict__ w, unsigned short* __restrict__ wB) {
  int gid = blockIdx.x * 256 + threadIdx.x;
  if (gid >= CIN * K3) return;
  int n = gid / K3, k = gid % K3;
  int t = k / CIN, i = k % CIN;                // k = tap*512 + cin
  wB[gid] = f2bf(w[n * (CIN * 9) + i * 9 + t]);
}

__global__ void pack_whead_k(const float* __restrict__ wcls, const float* __restrict__ wbox,
                             const float* __restrict__ bcls, const float* __restrict__ bbox,
                             unsigned short* __restrict__ w1, float* __restrict__ bias1) {
  int gid = blockIdx.x * 256 + threadIdx.x;
  if (gid < 64 * CIN) {
    int n = gid / CIN, c = gid % CIN;
    float v = 0.f;
    if (n < 9)       v = wcls[n * CIN + c];
    else if (n < 45) v = wbox[(n - 9) * CIN + c];
    w1[gid] = f2bf(v);
  }
  if (gid < 64) {
    float b = 0.f;
    if (gid < 9)       b = bcls[gid];
    else if (gid < 45) b = bbox[gid - 9];
    bias1[gid] = b;
  }
}

__global__ void zero_misc_k(unsigned* __restrict__ hist, unsigned* __restrict__ scal) {
  int i = blockIdx.x * 256 + threadIdx.x;
  if (i < 4096) hist[i] = 0u;
  if (i < 8)    scal[i] = 0u;
}

// ================= conv3x3 implicit GEMM (bf16 WMMA, async double-buffer) ===========
// C[M=10000, N=512] = im2col(featP) x W,  K=4608, 64x128 tile/WG, 32x32 tile/wave.
__global__ __launch_bounds__(256) void conv3x3_wmma_k(
    const unsigned short* __restrict__ featP,   // [102*102][512] bf16, zero halo
    const unsigned short* __restrict__ wB,      // [512][4608] bf16 (N-major)
    const float* __restrict__ bconv,
    unsigned short* __restrict__ rpnT)          // [HW][512] bf16
{
  __shared__ __align__(32) unsigned short As[2][64][32];
  __shared__ __align__(32) unsigned short Bs[2][128][32];

  const int tid  = threadIdx.x;
  const int lane = tid & 31;
  const int wave = tid >> 5;
  const int mBase = blockIdx.x * 64;
  const int nBase = blockIdx.y * 128;
  const int wm = (wave & 1) * 32;   // wave's M offset in tile
  const int wn = (wave >> 1) * 32;  // wave's N offset in tile
  const int r = lane & 15, half = lane >> 4;

  // A loader: 64 rows x 64B, 1 x 16B chunk per thread
  const int aRow = tid >> 2;
  const int aCol = (tid & 3) * 8;
  int m0 = mBase + aRow; if (m0 >= HW) m0 = HW - 1;   // clamp: tail rows unused
  const int ay = m0 / 100, ax = m0 % 100;
  const unsigned short* aBase = featP + ((ay + 1) * PADW + (ax + 1)) * CIN + aCol;

  // B loader: 128 rows x 64B, 2 x 16B chunks per thread
  const int bRow0 = tid >> 2;
  const int bOff  = (tid & 3) * 8;
  const unsigned short* bBase0 = wB + (size_t)(nBase + bRow0) * K3 + bOff;
  const unsigned short* bBase1 = bBase0 + (size_t)64 * K3;

  auto issue_stage = [&](int kc, int buf) {
    int tap = kc >> 4;               // 0..8
    int i0  = (kc & 15) * 32;        // channel base within tap
    int ky = tap / 3 - 1, kx = tap % 3 - 1;
    int aoff = (ky * PADW + kx) * CIN + i0;      // uniform per stage
    async_copy16(aBase + aoff,      &As[buf][aRow][aCol]);
    async_copy16(bBase0 + kc * 32,  &Bs[buf][bRow0][bOff]);
    async_copy16(bBase1 + kc * 32,  &Bs[buf][bRow0 + 64][bOff]);
  };

  v8f acc[2][2] = {};
  issue_stage(0, 0);

  for (int kc = 0; kc < KSTEPS; ++kc) {
    const int buf = kc & 1;
    if (kc + 1 < KSTEPS) {
      issue_stage(kc + 1, buf ^ 1);
      wait_async<3>();               // stage kc resident (in-order completion)
    } else {
      wait_async<0>();
    }
    __syncthreads();

    Frag a0, a1, b0, b1;
    // A 16x32 bf16 per-lane layout: lanes 0-15 K{0..7,16..23}, lanes 16-31 K{8..15,24..31}
    a0.q[0] = *(const uint4*)&As[buf][wm + r][half * 8];
    a0.q[1] = *(const uint4*)&As[buf][wm + r][half * 8 + 16];
    a1.q[0] = *(const uint4*)&As[buf][wm + 16 + r][half * 8];
    a1.q[1] = *(const uint4*)&As[buf][wm + 16 + r][half * 8 + 16];
    // B 32x16 bf16 per-lane layout: lanes 0-15 K{0..7,16..23}, lanes 16-31 K{8..15,24..31}
    b0.q[0] = *(const uint4*)&Bs[buf][wn + r][half * 8];
    b0.q[1] = *(const uint4*)&Bs[buf][wn + r][half * 8 + 16];
    b1.q[0] = *(const uint4*)&Bs[buf][wn + 16 + r][half * 8];
    b1.q[1] = *(const uint4*)&Bs[buf][wn + 16 + r][half * 8 + 16];

    acc[0][0] = wmma_bf16(a0, b0, acc[0][0]);
    acc[0][1] = wmma_bf16(a0, b1, acc[0][1]);
    acc[1][0] = wmma_bf16(a1, b0, acc[1][0]);
    acc[1][1] = wmma_bf16(a1, b1, acc[1][1]);
    __syncthreads();
  }

#pragma unroll
  for (int mi = 0; mi < 2; ++mi)
#pragma unroll
    for (int ni = 0; ni < 2; ++ni) {
      int n = nBase + wn + ni * 16 + r;
      float bias = bconv[n];
#pragma unroll
      for (int i = 0; i < 8; ++i) {
        int mm = mBase + wm + mi * 16 + half * 8 + i;
        if (mm < HW) {
          float v = acc[mi][ni][i] + bias;
          v = v > 0.f ? v : 0.f;
          rpnT[mm * CIN + n] = f2bf(v);
        }
      }
    }
}

// ================= 1x1 heads as GEMM (N=64: 9 cls + 36 box + pad) =================
__global__ __launch_bounds__(256) void head_wmma_k(
    const unsigned short* __restrict__ rpnT,  // [HW][512] bf16
    const unsigned short* __restrict__ w1,    // [64][512] bf16
    const float* __restrict__ bias1,
    float* __restrict__ headf)                // [HW][45]
{
  __shared__ __align__(32) unsigned short As[32][32];
  __shared__ __align__(32) unsigned short Bs[64][32];

  const int tid  = threadIdx.x;
  const int lane = tid & 31;
  const int wave = tid >> 5;
  const int mBase = blockIdx.x * 32;
  const int mSub = (wave & 1) * 16;
  const int nSub = (wave >> 1) * 16;

  const bool isA = tid >= 128;
  int aRow = 0, aCol = 0, bRow = 0, bKoff = 0;
  if (isA) { int tt = tid - 128; aRow = tt >> 2; aCol = (tt & 3) * 8; }
  else     { bRow = tid >> 1; bKoff = (tid & 1) * 16; }

  v8f acc = {};
  const int r = lane & 15, half = lane >> 4;

  for (int kc = 0; kc < 16; ++kc) {
    if (isA) {
      int m = mBase + aRow;
      uint4 v = make_uint4(0u, 0u, 0u, 0u);
      if (m < HW) v = *(const uint4*)&rpnT[m * CIN + kc * 32 + aCol];
      *(uint4*)&As[aRow][aCol] = v;
    } else {
      const unsigned short* src = &w1[bRow * CIN + kc * 32 + bKoff];
      *(uint4*)&Bs[bRow][bKoff]     = *(const uint4*)src;
      *(uint4*)&Bs[bRow][bKoff + 8] = *(const uint4*)(src + 8);
    }
    __syncthreads();

    Frag a, b;
    a.q[0] = *(const uint4*)&As[mSub + r][half * 8];
    a.q[1] = *(const uint4*)&As[mSub + r][half * 8 + 16];
    b.q[0] = *(const uint4*)&Bs[nSub + r][half * 8];
    b.q[1] = *(const uint4*)&Bs[nSub + r][half * 8 + 16];
    acc = wmma_bf16(a, b, acc);
    __syncthreads();
  }

  int n = nSub + r;
  if (n < 45) {
    float bias = bias1[n];
    for (int i = 0; i < 8; ++i) {
      int m = mBase + mSub + half * 8 + i;
      if (m < HW) headf[m * 45 + n] = acc[i] + bias;
    }
  }
}

// ================= top-2000 select (radix-histogram) =================
__global__ void score_hist_k(const float* __restrict__ headf,
                             unsigned* __restrict__ keyArr, unsigned* __restrict__ hist) {
  int idx = blockIdx.x * 256 + threadIdx.x;
  if (idx >= NPROP) return;
  int p = idx / NANC, a = idx % NANC;
  unsigned u = __float_as_uint(headf[p * 45 + a]);   // raw logit: sigmoid is monotone
  u = (u & 0x80000000u) ? ~u : (u | 0x80000000u);    // order-preserving key
  keyArr[idx] = u;
  atomicAdd(&hist[u >> 20], 1u);
}

__global__ void find_threshold_k(const unsigned* __restrict__ hist, unsigned* __restrict__ scal) {
  if (threadIdx.x != 0 || blockIdx.x != 0) return;
  unsigned total = 0; unsigned bstar = 0;
  for (int b = 4095; b >= 0; --b) {
    total += hist[b];
    if (total >= PRE_K) { bstar = (unsigned)b; break; }
  }
  scal[0] = bstar;
}

__global__ void compact_cand_k(const unsigned* __restrict__ keyArr, unsigned* __restrict__ scal,
                               unsigned* __restrict__ candIdx, unsigned* __restrict__ candKey) {
  int idx = blockIdx.x * 256 + threadIdx.x;
  if (idx >= NPROP) return;
  unsigned u = keyArr[idx];
  if ((u >> 20) >= scal[0]) {
    unsigned pos = atomicAdd(&scal[1], 1u);
    candIdx[pos] = (unsigned)idx;
    candKey[pos] = u;
  }
}

__global__ void rank_top2000_k(const unsigned* __restrict__ scal,
                               const unsigned* __restrict__ candIdx,
                               const unsigned* __restrict__ candKey,
                               unsigned* __restrict__ topIdx) {
  int i = blockIdx.x * 256 + threadIdx.x;
  unsigned n = scal[1];
  if (i >= (int)n) return;
  unsigned key = candKey[i], myidx = candIdx[i];
  unsigned rank = 0;
  for (unsigned j = 0; j < n; ++j) {
    unsigned kj = candKey[j];
    rank += (kj > key) || (kj == key && candIdx[j] < myidx);   // lax.top_k tie-break
  }
  if (rank < PRE_K) topIdx[rank] = myidx;
}

// ================= decode + clip + valid + sigmoid for top-2000 =================
__global__ void decode_top_k(const unsigned* __restrict__ topIdx, const float* __restrict__ headf,
                             float* __restrict__ boxes, float* __restrict__ scores,
                             unsigned* __restrict__ validArr) {
  int i = blockIdx.x * 256 + threadIdx.x;
  if (i >= PRE_K) return;
  int g = (int)topIdx[i];
  int p = g / NANC, a = g % NANC;
  int y = p / 100, x = p % 100;
  int ridx = a / 3, sidx = a % 3;
  float ratio = (ridx == 0) ? 0.5f : (ridx == 1) ? 1.0f : 2.0f;
  float scale = (sidx == 0) ? 128.f : (sidx == 1) ? 256.f : 512.f;
  float hr = sqrtf(ratio), wr = 1.f / hr;
  float wsz = wr * scale, hsz = hr * scale;
  float sx = (float)x * 16.f, sy = (float)y * 16.f;
  float ax1 = sx + rintf(-0.5f * wsz), ay1 = sy + rintf(-0.5f * hsz);
  float ax2 = sx + rintf( 0.5f * wsz), ay2 = sy + rintf( 0.5f * hsz);
  float w = ax2 - ax1, h = ay2 - ay1;
  float cx = ax1 + 0.5f * w, cy = ay1 + 0.5f * h;
  const float* hp = &headf[p * 45];
  float dx = hp[9 + a * 4 + 0], dy = hp[9 + a * 4 + 1];
  float dw = hp[9 + a * 4 + 2], dh = hp[9 + a * 4 + 3];
  float pcx = dx * w + cx, pcy = dy * h + cy;
  float pw = expf(dw) * w,  ph = expf(dh) * h;
  float x1 = fminf(fmaxf(pcx - 0.5f * pw, 0.f), 1600.f);
  float y1 = fminf(fmaxf(pcy - 0.5f * ph, 0.f), 1600.f);
  float x2 = fminf(fmaxf(pcx + 0.5f * pw, 0.f), 1600.f);
  float y2 = fminf(fmaxf(pcy + 0.5f * ph, 0.f), 1600.f);
  boxes[i * 4 + 0] = x1; boxes[i * 4 + 1] = y1;
  boxes[i * 4 + 2] = x2; boxes[i * 4 + 3] = y2;
  scores[i] = 1.f / (1.f + expf(-hp[a]));
  validArr[i] = ((x2 - x1) >= 16.f && (y2 - y1) >= 16.f) ? 1u : 0u;
}

// ================= NMS: parallel IoU bitmask + sequential sweep =================
__global__ void nms_mask_k(const float* __restrict__ boxes, unsigned* __restrict__ mask) {
  int t = blockIdx.x * 256 + threadIdx.x;
  if (t >= PRE_K * NMSW) return;
  int i = t / NMSW, jb = t % NMSW;
  float ax1 = boxes[i * 4], ay1 = boxes[i * 4 + 1], ax2 = boxes[i * 4 + 2], ay2 = boxes[i * 4 + 3];
  float areaA = (ax2 - ax1) * (ay2 - ay1);
  unsigned bits = 0;
  for (int jj = 0; jj < 32; ++jj) {
    int j = jb * 32 + jj;
    if (j >= PRE_K) break;
    float bx1 = boxes[j * 4], by1 = boxes[j * 4 + 1], bx2 = boxes[j * 4 + 2], by2 = boxes[j * 4 + 3];
    float xx1 = fmaxf(ax1, bx1), yy1 = fmaxf(ay1, by1);
    float xx2 = fminf(ax2, bx2), yy2 = fminf(ay2, by2);
    float iw = fmaxf(xx2 - xx1, 0.f), ih = fmaxf(yy2 - yy1, 0.f);
    float inter = iw * ih;
    float iou = inter / (areaA + (bx2 - bx1) * (by2 - by1) - inter);
    if (iou > 0.7f) bits |= (1u << jj);
  }
  mask[i * NMSW + jb] = bits;
}

__global__ void nms_sweep_k(const unsigned* __restrict__ mask, const unsigned* __restrict__ validArr,
                            const float* __restrict__ scores, float* __restrict__ maskedS) {
  if (threadIdx.x != 0 || blockIdx.x != 0) return;
  unsigned kw[NMSW];
  for (int w = 0; w < NMSW; ++w) {
    unsigned word = 0;
    for (int b = 0; b < 32; ++b) {
      int j = w * 32 + b;
      if (j < PRE_K && validArr[j]) word |= (1u << b);
    }
    kw[w] = word;
  }
  for (int i = 0; i < PRE_K; ++i) {
    if (!((kw[i >> 5] >> (i & 31)) & 1u)) continue;
    int w0 = i >> 5;
    for (int w = w0; w < NMSW; ++w) {
      unsigned m = mask[i * NMSW + w];
      if (w == w0) {
        int s = (i & 31) + 1;
        m = (s >= 32) ? 0u : (m & (0xFFFFFFFFu << s));  // only suppress j > i
      }
      kw[w] &= ~m;
    }
  }
  for (int i = 0; i < PRE_K; ++i)
    maskedS[i] = ((kw[i >> 5] >> (i & 31)) & 1u) ? scores[i] : -1.0f;
}

// ================= final top-1000 (rank-based, stable) =================
__global__ void final_topk_k(const float* __restrict__ maskedS, const float* __restrict__ boxes,
                             float* __restrict__ out) {
  int i = blockIdx.x * 256 + threadIdx.x;
  if (i >= PRE_K) return;
  float s = maskedS[i];
  int rank = 0;
  for (int j = 0; j < PRE_K; ++j) {
    float sj = maskedS[j];
    rank += (sj > s) || (sj == s && j < i);
  }
  if (rank < POST_K) {
    out[rank * 4 + 0] = boxes[i * 4 + 0];
    out[rank * 4 + 1] = boxes[i * 4 + 1];
    out[rank * 4 + 2] = boxes[i * 4 + 2];
    out[rank * 4 + 3] = boxes[i * 4 + 3];
    out[4 * POST_K + rank] = s;
  }
}

// ================= host launch =================
extern "C" void kernel_launch(void* const* d_in, const int* in_sizes, int n_in,
                              void* d_out, int out_size, void* d_ws, size_t ws_size,
                              hipStream_t stream) {
  (void)in_sizes; (void)n_in; (void)out_size; (void)ws_size;
  const float* feat   = (const float*)d_in[1];
  const float* w_conv = (const float*)d_in[2];
  const float* b_conv = (const float*)d_in[3];
  const float* w_cls  = (const float*)d_in[4];
  const float* b_cls  = (const float*)d_in[5];
  const float* w_box  = (const float*)d_in[6];
  const float* b_box  = (const float*)d_in[7];
  float* out = (float*)d_out;

  char* ws = (char*)d_ws;
  size_t off = 0;
  auto alloc = [&](size_t bytes) -> void* {
    void* p = ws + off;
    off = (off + bytes + 255) & ~(size_t)255;
    return p;
  };
  unsigned short* featP   = (unsigned short*)alloc((size_t)PADW * PADW * CIN * 2);
  unsigned short* wB      = (unsigned short*)alloc((size_t)CIN * K3 * 2);
  unsigned short* rpnT    = (unsigned short*)alloc((size_t)HW * CIN * 2);
  unsigned short* w1      = (unsigned short*)alloc((size_t)64 * CIN * 2);
  float*          bias1   = (float*)alloc(64 * 4);
  float*          headf   = (float*)alloc((size_t)HW * 45 * 4);
  unsigned*       keyArr  = (unsigned*)alloc((size_t)NPROP * 4);
  unsigned*       hist    = (unsigned*)alloc(4096 * 4);
  unsigned*       scal    = (unsigned*)alloc(8 * 4);
  unsigned*       candIdx = (unsigned*)alloc((size_t)NPROP * 4);
  unsigned*       candKey = (unsigned*)alloc((size_t)NPROP * 4);
  unsigned*       topIdx  = (unsigned*)alloc(PRE_K * 4);
  float*          boxes   = (float*)alloc(PRE_K * 4 * 4);
  float*          scores  = (float*)alloc(PRE_K * 4);
  unsigned*       validA  = (unsigned*)alloc(PRE_K * 4);
  unsigned*       nmsMask = (unsigned*)alloc((size_t)PRE_K * NMSW * 4);
  float*          maskedS = (float*)alloc(PRE_K * 4);

  zero_misc_k<<<17, 256, 0, stream>>>(hist, scal);
  zero_pad_k<<<(PADW * PADW * CIN / 2 + 255) / 256, 256, 0, stream>>>((unsigned*)featP);
  pack_feat_k<<<(HW * CIN + 255) / 256, 256, 0, stream>>>(feat, featP);
  pack_wconv_k<<<(CIN * K3 + 255) / 256, 256, 0, stream>>>(w_conv, wB);
  pack_whead_k<<<(64 * CIN + 255) / 256, 256, 0, stream>>>(w_cls, w_box, b_cls, b_box, w1, bias1);

  conv3x3_wmma_k<<<dim3((HW + 63) / 64, CIN / 128), 256, 0, stream>>>(featP, wB, b_conv, rpnT);
  head_wmma_k<<<(HW + 31) / 32, 256, 0, stream>>>(rpnT, w1, bias1, headf);

  score_hist_k<<<(NPROP + 255) / 256, 256, 0, stream>>>(headf, keyArr, hist);
  find_threshold_k<<<1, 64, 0, stream>>>(hist, scal);
  compact_cand_k<<<(NPROP + 255) / 256, 256, 0, stream>>>(keyArr, scal, candIdx, candKey);
  rank_top2000_k<<<(NPROP + 255) / 256, 256, 0, stream>>>(scal, candIdx, candKey, topIdx);
  decode_top_k<<<(PRE_K + 255) / 256, 256, 0, stream>>>(topIdx, headf, boxes, scores, validA);
  nms_mask_k<<<(PRE_K * NMSW + 255) / 256, 256, 0, stream>>>(boxes, nmsMask);
  nms_sweep_k<<<1, 64, 0, stream>>>(nmsMask, validA, scores, maskedS);
  final_topk_k<<<(PRE_K + 255) / 256, 256, 0, stream>>>(maskedS, boxes, out);
}
